// GAT_L_8564164788536
// MI455X (gfx1250) — matrix operand order repaired and checked by reading, not compiled, not run on hardware
//
#include <hip/hip_runtime.h>
#include <hip/hip_bf16.h>
#include <math.h>

// ---------------- problem constants ----------------
#define NN   50000
#define EE   800000
#define HH   3
#define GG   64

typedef __attribute__((ext_vector_type(16))) _Float16 v16h;
typedef __attribute__((ext_vector_type(8)))  float    v8f;
typedef __attribute__((ext_vector_type(4)))  float    v4f;

static inline int cdiv(long long a, long long b) { return (int)((a + b - 1) / b); }

// ---------------- fill kernels ----------------
__global__ void fill_f(float* p, float v, long long n) {
    long long i = (long long)blockIdx.x * blockDim.x + threadIdx.x;
    if (i < n) p[i] = v;
}
__global__ void fill_i(int* p, int v, long long n) {
    long long i = (long long)blockIdx.x * blockDim.x + threadIdx.x;
    if (i < n) p[i] = v;
}

// ---------------- WMMA GEMM: C[M,N] = A[M,K] * B[K,N], f32 io, f16 math ----
// block = 160 threads = 5 waves; each wave owns one 16-row M tile.
// grid.x = 625 (M = 50000 = 625*5*16), grid.y strips of 16*NT columns at nbase.
// All column ranges are launched exactly in-bounds (no guards, no divergence).
template <int NT>
__global__ __launch_bounds__(160) void gemm_wmma(const float* __restrict__ A,
                                                 const float* __restrict__ B,
                                                 float* __restrict__ C,
                                                 int K, int N, int nbase) {
    // Fragment-ordered B tile: NT subtiles x 32 lanes x 16 halves (32B per lane)
    __shared__ alignas(32) _Float16 smB[NT * 32 * 16];

    const int lane = threadIdx.x & 31;
    const int wave = threadIdx.x >> 5;
    const int mt   = blockIdx.x * 5 + wave;          // always valid
    const int n0   = nbase + blockIdx.y * (16 * NT);

    const bool hi  = lane >= 16;
    const int  m   = lane & 15;
    const int  row = mt * 16 + m;

    v8f acc[NT];
    #pragma unroll
    for (int t = 0; t < NT; ++t) acc[t] = (v8f){};

    for (int kb = 0; kb < K; kb += 32) {
        // ---- cooperative fragment-ordered B tile fill ----
        // thread p < 32*NT produces fragment row (t = p>>5, ln = p&31):
        //   halves j = B[kb + (ln>=16?16:0) + j][n0 + t*16 + (ln&15)]
        if (threadIdx.x < 32 * NT) {
            const int t  = threadIdx.x >> 5;
            const int ln = threadIdx.x & 31;
            const int n  = n0 + t * 16 + (ln & 15);
            const float* bp = B + (size_t)(kb + ((ln >= 16) ? 16 : 0)) * N + n;
            v16h tmp;
            #pragma unroll
            for (int j = 0; j < 16; ++j) tmp[j] = (_Float16)bp[(size_t)j * N];
            ((v16h*)smB)[threadIdx.x] = tmp;
        }
        __syncthreads();

        // ---- A fragment: lanes 0-15: K kb+0..15; lanes 16-31: K kb+16..31 ----
        const v4f* ap = (const v4f*)(A + (size_t)row * K + kb + (hi ? 16 : 0));
        v4f q0 = ap[0], q1 = ap[1], q2 = ap[2], q3 = ap[3];
        v16h a;
        #pragma unroll
        for (int i = 0; i < 4; ++i) {
            a[i]      = (_Float16)q0[i];
            a[i + 4]  = (_Float16)q1[i];
            a[i + 8]  = (_Float16)q2[i];
            a[i + 12] = (_Float16)q3[i];
        }

        #pragma unroll
        for (int t = 0; t < NT; ++t) {
            v16h b = ((const v16h*)smB)[t * 32 + lane];   // one 32B LDS load
            acc[t] = __builtin_amdgcn_wmma_f32_16x16x32_f16(false, a, false, b,
                                                            (short)0, acc[t], false, false);
        }
        __syncthreads();
    }

    // ---- store D: VGPR r holds (M = r + (hi?8:0), N = lane&15) ----
    const int mrow0 = mt * 16 + (hi ? 8 : 0);
    #pragma unroll
    for (int t = 0; t < NT; ++t) {
        const int n = n0 + t * 16 + m;
        #pragma unroll
        for (int r = 0; r < 8; ++r)
            C[(size_t)(mrow0 + r) * N + n] = acc[t][r];
    }
}

// ---------------- degree + edge-attr sum over destinations ----------------
__global__ void deg_easum_k(const int* __restrict__ dst, const float* __restrict__ ea,
                            float* deg, float* easum) {
    int e = blockIdx.x * blockDim.x + threadIdx.x;
    if (e >= EE) return;
    int d = dst[e];
    atomicAdd(&deg[d], 1.0f);
    atomicAdd(&easum[d], ea[e]);
}
__global__ void mean_ea_k(const float* easum, const float* deg, float* mean_ea) {
    int i = blockIdx.x * blockDim.x + threadIdx.x;
    if (i >= NN) return;
    mean_ea[i] = easum[i] / fmaxf(deg[i], 1.0f);
}

// ---------------- coef[h] = sum_c We[h,c]*a_e[h,c] (rank-1 edge transform) ---
__global__ void coef_k(const float* __restrict__ We, const float* __restrict__ aew,
                       float* coef, int C) {
    int t = threadIdx.x;
    for (int h = 0; h < HH; ++h) {
        float s = 0.0f;
        for (int c = t; c < C; c += blockDim.x) s += We[h * C + c] * aew[h * C + c];
        atomicAdd(&coef[h], s);
    }
}

// ---------------- per-node attention scalars (wave per node) --------------
__global__ void attn_node_k(const float* __restrict__ xs,
                            const float* __restrict__ a_s, const float* __restrict__ a_d,
                            float* asrc, float* adst, int C) {
    int wv = (blockIdx.x * blockDim.x + threadIdx.x) >> 5;
    int lane = threadIdx.x & 31;
    if (wv >= NN) return;
    for (int h = 0; h < HH; ++h) {
        float ss = 0.0f, sd = 0.0f;
        const float* xp = xs + (size_t)wv * HH * C + h * C;
        for (int c = lane; c < C; c += 32) {
            float v = xp[c];
            ss += v * a_s[h * C + c];
            sd += v * a_d[h * C + c];
        }
        #pragma unroll
        for (int o = 16; o > 0; o >>= 1) {
            ss += __shfl_xor(ss, o, 32);
            sd += __shfl_xor(sd, o, 32);
        }
        if (lane == 0) { asrc[wv * HH + h] = ss; adst[wv * HH + h] = sd; }
    }
}

// ---------------- ordered-int float max helpers ----------------
__device__ __forceinline__ int f2ord(float x) {
    int i = __float_as_int(x);
    return i >= 0 ? i : (i ^ 0x7fffffff);
}
__device__ __forceinline__ float ord2f(int i) {
    return __int_as_float(i >= 0 ? i : (i ^ 0x7fffffff));
}

// ---------------- pass 1: alpha = leaky_relu, scatter-max per (dst,h) -----
__global__ void alpha_max_k(const int* __restrict__ src, const int* __restrict__ dst,
                            const float* __restrict__ ea, const float* __restrict__ mean_ea,
                            const float* __restrict__ asrc, const float* __restrict__ adst,
                            const float* __restrict__ coef,
                            float* __restrict__ alpha, int* __restrict__ mxI) {
    int e = blockIdx.x * blockDim.x + threadIdx.x;
    if (e >= EE + NN) return;
    int s, d; float eav;
    if (e < EE) { s = src[e]; d = dst[e]; eav = ea[e]; }
    else        { s = d = e - EE; eav = mean_ea[s]; }
    #pragma unroll
    for (int h = 0; h < HH; ++h) {
        float a = asrc[s * HH + h] + adst[d * HH + h] + eav * coef[h];
        a = (a > 0.0f) ? a : 0.2f * a;
        alpha[(size_t)e * HH + h] = a;
        atomicMax(&mxI[d * HH + h], f2ord(a));
    }
}

// ---------------- pass 2: exp(alpha - max), scatter-sum denominator --------
__global__ void expden_k(const int* __restrict__ dst, float* __restrict__ alpha,
                         const int* __restrict__ mxI, float* __restrict__ den) {
    int e = blockIdx.x * blockDim.x + threadIdx.x;
    if (e >= EE + NN) return;
    int d = (e < EE) ? dst[e] : (e - EE);
    #pragma unroll
    for (int h = 0; h < HH; ++h) {
        float ex = __expf(alpha[(size_t)e * HH + h] - ord2f(mxI[d * HH + h]));
        alpha[(size_t)e * HH + h] = ex;
        atomicAdd(&den[d * HH + h], ex);
    }
}

// ---------------- pass 3: out[d,f..f+3] += xs[s,f..f+3] * w ----------------
__global__ void agg_k(const int* __restrict__ src, const int* __restrict__ dst,
                      const float* __restrict__ xs, const float* __restrict__ alpha,
                      const float* __restrict__ den, float* __restrict__ out,
                      int HC, int C) {
    const int HC4 = HC >> 2;
    long long tid = (long long)blockIdx.x * blockDim.x + threadIdx.x;
    long long total = (long long)(EE + NN) * HC4;
    if (tid >= total) return;
    int e  = (int)(tid / HC4);
    int f  = (int)(tid % HC4) << 2;          // C is a multiple of 4 -> same head
    int s  = (e < EE) ? src[e] : (e - EE);
    int d  = (e < EE) ? dst[e] : (e - EE);
    int h  = f / C;
    float w = alpha[(size_t)e * HH + h] / den[d * HH + h];
    v4f xv = *(const v4f*)(xs + (size_t)s * HC + f);
    float* op = out + (size_t)d * HC + f;
    atomicAdd(op + 0, xv[0] * w);
    atomicAdd(op + 1, xv[1] * w);
    atomicAdd(op + 2, xv[2] * w);
    atomicAdd(op + 3, xv[3] * w);
}

// ---------------- bias + relu ----------------
__global__ void bias_relu_k(float* y, const float* __restrict__ b, long long n, int HC) {
    long long i = (long long)blockIdx.x * blockDim.x + threadIdx.x;
    if (i >= n) return;
    y[i] = fmaxf(y[i] + b[(int)(i % HC)], 0.0f);
}

// ---------------- BatchNorm: LDS-staged column sums ----------------
__global__ void bn_stats_k(const float* __restrict__ y, float* sums, float* sumsq,
                           int C, int rowsPerBlock) {
    extern __shared__ float sm[];                 // 2*C floats
    for (int i = threadIdx.x; i < 2 * C; i += blockDim.x) sm[i] = 0.0f;
    __syncthreads();
    long long r0 = (long long)blockIdx.x * rowsPerBlock;
    long long total = (long long)rowsPerBlock * C;
    for (long long i = threadIdx.x; i < total; i += blockDim.x) {
        long long r = r0 + i / C;
        int c = (int)(i % C);
        if (r < NN) {
            float v = y[r * C + c];
            atomicAdd(&sm[c], v);
            atomicAdd(&sm[C + c], v * v);
        }
    }
    __syncthreads();
    for (int c = threadIdx.x; c < C; c += blockDim.x) {
        atomicAdd(&sums[c], sm[c]);
        atomicAdd(&sumsq[c], sm[C + c]);
    }
}
__global__ void bn_fin_k(const float* sums, const float* sumsq,
                         const float* g, const float* be,
                         float* scale, float* shift, int C) {
    int c = blockIdx.x * blockDim.x + threadIdx.x;
    if (c >= C) return;
    const float invN = 1.0f / (float)NN;
    float m = sums[c] * invN;
    float v = sumsq[c] * invN - m * m;
    float sc = g[c] * rsqrtf(v + 1e-5f);
    scale[c] = sc;
    shift[c] = be[c] - m * sc;
}
__global__ void bn_apply_k(float* y, const float* __restrict__ scale,
                           const float* __restrict__ shift, long long n, int C) {
    long long i = (long long)blockIdx.x * blockDim.x + threadIdx.x;
    if (i >= n) return;
    int c = (int)(i % C);
    y[i] = y[i] * scale[c] + shift[c];
}

// ---------------- global add pool ----------------
__global__ void pool_k(const float* __restrict__ y, const int* __restrict__ batch,
                       float* pooled, int C) {
    long long i = (long long)blockIdx.x * blockDim.x + threadIdx.x;
    if (i >= (long long)NN * C) return;
    int n = (int)(i / C), c = (int)(i % C);
    atomicAdd(&pooled[batch[n] * C + c], y[i]);
}

// ---------------- tiny FC head ----------------
__global__ void fc_k(const float* __restrict__ in, const float* __restrict__ W,
                     const float* __restrict__ b, float* out,
                     int rows, int K, int Nc, int do_relu) {
    int i = blockIdx.x * blockDim.x + threadIdx.x;
    if (i >= rows * Nc) return;
    int r = i / Nc, c = i % Nc;
    float s = b[c];
    for (int k = 0; k < K; ++k) s += in[r * K + k] * W[k * Nc + c];
    out[i] = do_relu ? fmaxf(s, 0.0f) : s;
}

// =======================================================================
extern "C" void kernel_launch(void* const* d_in, const int* in_sizes, int n_in,
                              void* d_out, int out_size, void* d_ws, size_t ws_size,
                              hipStream_t stream) {
    const float* x     = (const float*)d_in[0];
    const int*   ei    = (const int*)d_in[1];
    const int*   src   = ei;
    const int*   dst   = ei + EE;
    const float* ea    = (const float*)d_in[2];
    const int*   batch = (const int*)d_in[3];

    struct LayerP { const float *W, *as, *ad, *We, *ae, *b, *g, *be; int K, C, HC; };
    auto getL = [&](int base, int K, int C) -> LayerP {
        LayerP p;
        p.W  = (const float*)d_in[base + 0];
        p.as = (const float*)d_in[base + 1];
        p.ad = (const float*)d_in[base + 2];
        p.We = (const float*)d_in[base + 3];
        p.ae = (const float*)d_in[base + 4];
        p.b  = (const float*)d_in[base + 5];
        p.g  = (const float*)d_in[base + 6];
        p.be = (const float*)d_in[base + 7];
        p.K = K; p.C = C; p.HC = HH * C;
        return p;
    };
    LayerP L1 = getL(4, 128, 128);
    LayerP L2 = getL(12, 384, 64);
    LayerP L3 = getL(20, 192, 32);
    const float* fc1W = (const float*)d_in[28]; const float* fc1b = (const float*)d_in[29];
    const float* fc2W = (const float*)d_in[30]; const float* fc2b = (const float*)d_in[31];
    const float* fc3W = (const float*)d_in[32]; const float* fc3b = (const float*)d_in[33];

    // -------- workspace bump allocator --------
    float* wsf = (float*)d_ws;
    size_t off = 0;
    auto nxt = [&](size_t n) { float* p = wsf + off; off += n; return p; };

    float* bufA   = nxt((size_t)NN * 384);   // xs (transformed features)
    float* bufB   = nxt((size_t)NN * 384);   // layer 1 & 3 outputs
    float* bufC   = nxt((size_t)NN * 192);   // layer 2 output
    float* asrc   = nxt((size_t)NN * HH);
    float* adst   = nxt((size_t)NN * HH);
    float* deg    = nxt(NN);
    float* easum  = nxt(NN);
    float* meanea = nxt(NN);
    float* den    = nxt((size_t)NN * HH);
    float* alphaw = nxt((size_t)(EE + NN) * HH);
    int*   mxI    = (int*)nxt((size_t)NN * HH);
    float* coef   = nxt(16);
    float* sums   = nxt(384);
    float* sumsq  = nxt(384);
    float* scale  = nxt(384);
    float* shift  = nxt(384);
    float* pooled = nxt((size_t)GG * 96);
    float* h1     = nxt((size_t)GG * 128);
    float* h2     = nxt((size_t)GG * 64);

    const int T = 256;
    const int E2 = EE + NN;

    // -------- one-time: degrees + mean incoming edge attr --------
    fill_f<<<cdiv(2 * NN, T), T, 0, stream>>>(deg, 0.0f, 2 * NN);   // deg + easum contiguous
    deg_easum_k<<<cdiv(EE, T), T, 0, stream>>>(dst, ea, deg, easum);
    mean_ea_k<<<cdiv(NN, T), T, 0, stream>>>(easum, deg, meanea);

    // -------- per-layer driver --------
    auto run_layer = [&](const LayerP& L, const float* in, float* out) {
        const int C = L.C, HC = L.HC, K = L.K;
        const long long nelem = (long long)NN * HC;

        // 1) xs = in @ W  (WMMA): full 64-col strips, then the 32-col tail (HC=96)
        const int full = HC / 64;
        if (full > 0)
            gemm_wmma<4><<<dim3(625, full), 160, 0, stream>>>(in, L.W, bufA, K, HC, 0);
        if (HC % 64)
            gemm_wmma<2><<<dim3(625, 1), 160, 0, stream>>>(in, L.W, bufA, K, HC, full * 64);

        // 2) rank-1 edge coefficient
        fill_f<<<1, 16, 0, stream>>>(coef, 0.0f, HH);
        coef_k<<<1, 128, 0, stream>>>(L.We, L.ae, coef, C);

        // 3) per-node attention scalars
        attn_node_k<<<cdiv((long long)NN * 32, T), T, 0, stream>>>(bufA, L.as, L.ad, asrc, adst, C);

        // 4) scatter softmax
        fill_i<<<cdiv(NN * HH, T), T, 0, stream>>>(mxI, (int)0x80000000, NN * HH);
        fill_f<<<cdiv(NN * HH, T), T, 0, stream>>>(den, 0.0f, NN * HH);
        alpha_max_k<<<cdiv(E2, T), T, 0, stream>>>(src, dst, ea, meanea, asrc, adst, coef,
                                                   alphaw, mxI);
        expden_k<<<cdiv(E2, T), T, 0, stream>>>(dst, alphaw, mxI, den);

        // 5) weighted aggregation (vectorized: 4 features per thread)
        fill_f<<<cdiv(nelem, T), T, 0, stream>>>(out, 0.0f, nelem);
        agg_k<<<cdiv((long long)E2 * (HC >> 2), T), T, 0, stream>>>(src, dst, bufA, alphaw,
                                                                    den, out, HC, C);

        // 6) bias + relu
        bias_relu_k<<<cdiv(nelem, T), T, 0, stream>>>(out, L.b, nelem, HC);

        // 7) batchnorm
        fill_f<<<cdiv(2 * 384, T), T, 0, stream>>>(sums, 0.0f, 2 * 384);  // sums+sumsq contiguous
        const int RPB = 256;
        bn_stats_k<<<cdiv(NN, RPB), T, 2 * HC * (int)sizeof(float), stream>>>(out, sums, sumsq, HC, RPB);
        bn_fin_k<<<cdiv(HC, 128), 128, 0, stream>>>(sums, sumsq, L.g, L.be, scale, shift, HC);
        bn_apply_k<<<cdiv(nelem, T), T, 0, stream>>>(out, scale, shift, nelem, HC);
    };

    run_layer(L1, x,    bufB);   // [N,384]
    run_layer(L2, bufB, bufC);   // [N,192]
    run_layer(L3, bufC, bufB);   // [N, 96]

    // -------- pool + MLP head --------
    fill_f<<<cdiv(GG * 96, T), T, 0, stream>>>(pooled, 0.0f, GG * 96);
    pool_k<<<cdiv((long long)NN * 96, T), T, 0, stream>>>(bufB, batch, pooled, 96);

    fc_k<<<cdiv(GG * 128, T), T, 0, stream>>>(pooled, fc1W, fc1b, h1, GG, 96, 128, 1);
    fc_k<<<cdiv(GG * 64, T), T, 0, stream>>>(h1, fc2W, fc2b, h2, GG, 128, 64, 1);
    fc_k<<<cdiv(GG * 2, T), T, 0, stream>>>(h2, fc3W, fc3b, (float*)d_out, GG, 64, 2, 0);
}
